// GGNN_90013924589708
// MI455X (gfx1250) — matrix operand order repaired and checked by reading, not compiled, not run on hardware
//
#include <hip/hip_runtime.h>
#include <hip/hip_bf16.h>
#include <math.h>

// GGNN on MI455X (gfx1250). Dense matmuls via v_wmma_f32_16x16x32_bf16 with
// fp32 accumulate; A fragments held in VGPRs and reused across output tiles.
// Edge gather/scatter: coalesced b128 loads + f32 global atomics (working set
// ~166 MB fits the 192 MB L2). Scratch carve-out ~205 MB in d_ws.

typedef __bf16 bf16_t;
typedef __attribute__((ext_vector_type(16))) __bf16 v16bf;
typedef __attribute__((ext_vector_type(8)))  float  v8f;

#define IN_DIM  64
#define OUT_DIM 128
#define NT      4
#define NSTEPS  8

// ---------------- WMMA fragment loaders (per CDNA5 ISA 7.12.2) -------------

// A-matrix 16x32 bf16, row-major source with stride OUT_DIM.
// lane L (m = L&15, hi = L>>4), half i (vgpr v=i>>1):
//   K = (hi?8:0) + (v>=4?16:0) + 2*(v&3) + (i&1)
__device__ __forceinline__ v16bf load_A_bf(const bf16_t* H, int row0, int k0, int lane) {
  int m = lane & 15, hi = lane >> 4;
  const bf16_t* p = H + (size_t)(row0 + m) * OUT_DIM + k0;
  v16bf a;
#pragma unroll
  for (int i = 0; i < 16; ++i) {
    int v = i >> 1;
    int k = (hi ? 8 : 0) + ((v & 4) ? 16 : 0) + 2 * (v & 3) + (i & 1);
    a[i] = p[k];
  }
  return a;
}

__device__ __forceinline__ v16bf load_A_f32(const float* H, int row0, int k0, int lane) {
  int m = lane & 15, hi = lane >> 4;
  const float* p = H + (size_t)(row0 + m) * OUT_DIM + k0;
  v16bf a;
#pragma unroll
  for (int i = 0; i < 16; ++i) {
    int v = i >> 1;
    int k = (hi ? 8 : 0) + ((v & 4) ? 16 : 0) + 2 * (v & 3) + (i & 1);
    a[i] = (bf16_t)p[k];
  }
  return a;
}

// B-matrix 32x16 (KxN) with B[k][n] = W[j0+n][k0+k], W row-major stride OUT_DIM.
// lane L holds column n = L&15; halves i -> K = 8*(L>>4) + (i<8 ? i : i+8).
__device__ __forceinline__ v16bf load_B_bf(const bf16_t* W, int j0, int k0, int lane) {
  int n = lane & 15, hi = lane >> 4;
  const bf16_t* p = W + (size_t)(j0 + n) * OUT_DIM + k0 + (hi ? 8 : 0);
  v16bf b;
#pragma unroll
  for (int i = 0; i < 16; ++i) b[i] = p[i < 8 ? i : i + 8];
  return b;
}

__device__ __forceinline__ v8f wmma_bf16(v16bf a, v16bf b, v8f c) {
  return __builtin_amdgcn_wmma_f32_16x16x32_bf16(false, a, false, b, (short)0, c,
                                                 false, false);
}

// ---------------- utility kernels ------------------------------------------

__global__ void cvt_bf16_kernel(const float* __restrict__ src,
                                bf16_t* __restrict__ dst, int n) {
  int i = blockIdx.x * blockDim.x + threadIdx.x;
  if (i < n) dst[i] = (bf16_t)src[i];
}

__global__ void zero_kernel(float* __restrict__ p, size_t n) {
  size_t i = (size_t)blockIdx.x * blockDim.x + threadIdx.x;
  if (i < n) p[i] = 0.0f;
}

// h = [node_features | 0], also bf16 copy
__global__ void init_h_kernel(const float* __restrict__ nf,
                              float* __restrict__ h, bf16_t* __restrict__ hb, int N) {
  size_t i = (size_t)blockIdx.x * blockDim.x + threadIdx.x;
  if (i >= (size_t)N * OUT_DIM) return;
  int n = (int)(i >> 7), d = (int)(i & 127);
  float v = (d < IN_DIM) ? nf[(size_t)n * IN_DIM + d] : 0.0f;
  h[i] = v;
  hb[i] = (bf16_t)v;
}

// ---------------- per-step kernels -----------------------------------------

// Wh[t] = h @ Ws[t]^T + bs[t]
// Wave per (row-tile, etype); A fragments resident, sweep 8 col tiles (32 WMMA).
__global__ void transform_kernel(const bf16_t* __restrict__ Hb,
                                 const bf16_t* __restrict__ Wsb,
                                 const float* __restrict__ bs,
                                 float* __restrict__ Wh, int N, int nRowTiles) {
  int gid = blockIdx.x * blockDim.x + threadIdx.x;
  int wave = gid >> 5, lane = gid & 31;
  int t = wave & 3;            // edge type
  int rt = wave >> 2;          // row tile
  if (rt >= nRowTiles) return;
  int row0 = rt * 16;

  v16bf Aa[4];
#pragma unroll
  for (int kc = 0; kc < 4; ++kc) Aa[kc] = load_A_bf(Hb, row0, kc * 32, lane);

  const bf16_t* W = Wsb + (size_t)t * OUT_DIM * OUT_DIM;
  int nn = lane & 15, hi = lane >> 4;
  float* obase = Wh + ((size_t)t * N + row0) * OUT_DIM;

#pragma unroll
  for (int j = 0; j < 8; ++j) {
    v8f acc = {};
#pragma unroll
    for (int kc = 0; kc < 4; ++kc)
      acc = wmma_bf16(Aa[kc], load_B_bf(W, j * 16, kc * 32, lane), acc);
    int col = j * 16 + nn;
    float bias = bs[t * OUT_DIM + col];
    float* o = obase + col;
#pragma unroll
    for (int r = 0; r < 8; ++r)
      o[(size_t)(r + 8 * hi) * OUT_DIM] = acc[r] + bias;
  }
}

// a[dst] += Wh[etype][src]   (wave per edge; one b128 load + 4 f32 atomics/lane)
__global__ void scatter_kernel(const float* __restrict__ Wh,
                               const int* __restrict__ src,
                               const int* __restrict__ dst,
                               const int* __restrict__ et,
                               float* __restrict__ A, int E, int N) {
  int gid = blockIdx.x * blockDim.x + threadIdx.x;
  int wave = gid >> 5, lane = gid & 31;
  if (wave >= E) return;
  int s = src[wave], d = dst[wave], t = et[wave];
  const float4* m = (const float4*)(Wh + ((size_t)t * N + s) * OUT_DIM);
  float4 v = m[lane];
  float* o = A + (size_t)d * OUT_DIM + lane * 4;
  atomicAdd(o + 0, v.x);
  atomicAdd(o + 1, v.y);
  atomicAdd(o + 2, v.z);
  atomicAdd(o + 3, v.w);
}

// GRU cell: wave per (row tile, half); Aa/Ah resident, sweep 4 col tiles
// (96 WMMAs per wave) with fused sigmoid/tanh/blend.
__global__ void gru_kernel(const float* __restrict__ A,       // aggregated msgs (f32)
                           const float* __restrict__ Hf,      // old h f32
                           const bf16_t* __restrict__ Hb,     // old h bf16
                           const bf16_t* __restrict__ Wih,
                           const bf16_t* __restrict__ Whh,
                           const float* __restrict__ bih,
                           const float* __restrict__ bhh,
                           float* __restrict__ Hnf, bf16_t* __restrict__ Hnb,
                           int nRowTiles) {
  int gid = blockIdx.x * blockDim.x + threadIdx.x;
  int wave = gid >> 5, lane = gid & 31;
  int jh = wave & 1, rt = wave >> 1;
  if (rt >= nRowTiles) return;
  int row0 = rt * 16;

  v16bf Aa[4], Ah[4];
#pragma unroll
  for (int kc = 0; kc < 4; ++kc) {
    Aa[kc] = load_A_f32(A, row0, kc * 32, lane);
    Ah[kc] = load_A_bf(Hb, row0, kc * 32, lane);
  }

  int nn = lane & 15, hi = lane >> 4;

#pragma unroll
  for (int jj = 0; jj < 4; ++jj) {
    int j = jh * 4 + jj;
    v8f gir = {}, giz = {}, gin = {}, ghr = {}, ghz = {}, ghn = {};
#pragma unroll
    for (int kc = 0; kc < 4; ++kc) {
      int k0 = kc * 32;
      gir = wmma_bf16(Aa[kc], load_B_bf(Wih, 0   + j * 16, k0, lane), gir);
      giz = wmma_bf16(Aa[kc], load_B_bf(Wih, 128 + j * 16, k0, lane), giz);
      gin = wmma_bf16(Aa[kc], load_B_bf(Wih, 256 + j * 16, k0, lane), gin);
      ghr = wmma_bf16(Ah[kc], load_B_bf(Whh, 0   + j * 16, k0, lane), ghr);
      ghz = wmma_bf16(Ah[kc], load_B_bf(Whh, 128 + j * 16, k0, lane), ghz);
      ghn = wmma_bf16(Ah[kc], load_B_bf(Whh, 256 + j * 16, k0, lane), ghn);
    }
    int col = j * 16 + nn;
    float bir = bih[col], biz = bih[128 + col], bin = bih[256 + col];
    float bhr = bhh[col], bhz = bhh[128 + col], bhn = bhh[256 + col];
#pragma unroll
    for (int r = 0; r < 8; ++r) {
      int row = row0 + r + 8 * hi;
      size_t idx = (size_t)row * OUT_DIM + col;
      float rg = 1.0f / (1.0f + __expf(-((gir[r] + bir) + (ghr[r] + bhr))));
      float zg = 1.0f / (1.0f + __expf(-((giz[r] + biz) + (ghz[r] + bhz))));
      float ng = tanhf((gin[r] + bin) + rg * (ghn[r] + bhn));
      float hv = Hf[idx];
      float hn2 = (1.0f - zg) * ng + zg * hv;
      Hnf[idx] = hn2;
      Hnb[idx] = (bf16_t)hn2;
    }
  }
}

// ---------------- readout ---------------------------------------------------

__global__ void readout_acc_kernel(const float* __restrict__ H,
                                   const float* __restrict__ nf,
                                   const int* __restrict__ gids,
                                   float* __restrict__ feats, int N) {
  int gid = blockIdx.x * blockDim.x + threadIdx.x;
  int wave = gid >> 5, lane = gid & 31;
  if (wave >= N) return;
  int g = gids[wave];
  float* f = feats + (size_t)g * (OUT_DIM + IN_DIM);
  for (int c = lane; c < OUT_DIM + IN_DIM; c += 32) {
    float v = (c < OUT_DIM) ? H[(size_t)wave * OUT_DIM + c]
                            : nf[(size_t)wave * IN_DIM + (c - OUT_DIM)];
    atomicAdd(&f[c], v);
  }
}

__global__ void readout_final_kernel(const float* __restrict__ feats,
                                     const float* __restrict__ Wcls,
                                     const float* __restrict__ bcls,
                                     float* __restrict__ out, int G) {
  int gid = blockIdx.x * blockDim.x + threadIdx.x;
  int wave = gid >> 5, lane = gid & 31;
  if (wave >= G) return;
  float s = 0.0f;
  for (int c = lane; c < OUT_DIM + IN_DIM; c += 32)
    s += feats[(size_t)wave * (OUT_DIM + IN_DIM) + c] * Wcls[c];
#pragma unroll
  for (int off = 16; off > 0; off >>= 1) s += __shfl_xor(s, off, 32);
  if (lane == 0) out[wave] = 1.0f / (1.0f + __expf(-(s + bcls[0])));
}

// ---------------- launcher --------------------------------------------------

extern "C" void kernel_launch(void* const* d_in, const int* in_sizes, int n_in,
                              void* d_out, int out_size, void* d_ws, size_t ws_size,
                              hipStream_t stream) {
  const float* nf    = (const float*)d_in[0];
  const int*   src   = (const int*)d_in[1];
  const int*   dst   = (const int*)d_in[2];
  const int*   etype = (const int*)d_in[3];
  const int*   gids  = (const int*)d_in[4];
  const float* Ws    = (const float*)d_in[5];
  const float* bs    = (const float*)d_in[6];
  const float* Wih   = (const float*)d_in[7];
  const float* Whh   = (const float*)d_in[8];
  const float* bih   = (const float*)d_in[9];
  const float* bhh   = (const float*)d_in[10];
  const float* Wcls  = (const float*)d_in[11];
  const float* bcls  = (const float*)d_in[12];
  float* out = (float*)d_out;

  const int N = in_sizes[0] / IN_DIM;   // 50000
  const int E = in_sizes[1];            // 400000
  const int G = out_size;               // 64
  const int nRowTiles = N / 16;         // 3125 (N % 16 == 0)

  // scratch carve-out
  char* p = (char*)d_ws;
  auto alloc = [&](size_t bytes) -> char* {
    char* r = p;
    p += (bytes + 255) & ~(size_t)255;
    return r;
  };
  float*  hA    = (float*)alloc((size_t)N * OUT_DIM * sizeof(float));
  float*  hB    = (float*)alloc((size_t)N * OUT_DIM * sizeof(float));
  float*  aBuf  = (float*)alloc((size_t)N * OUT_DIM * sizeof(float));
  float*  WhBuf = (float*)alloc((size_t)NT * N * OUT_DIM * sizeof(float));
  bf16_t* hAb   = (bf16_t*)alloc((size_t)N * OUT_DIM * sizeof(bf16_t));
  bf16_t* hBb   = (bf16_t*)alloc((size_t)N * OUT_DIM * sizeof(bf16_t));
  bf16_t* Wsb   = (bf16_t*)alloc((size_t)NT * OUT_DIM * OUT_DIM * sizeof(bf16_t));
  bf16_t* Wihb  = (bf16_t*)alloc((size_t)3 * OUT_DIM * OUT_DIM * sizeof(bf16_t));
  bf16_t* Whhb  = (bf16_t*)alloc((size_t)3 * OUT_DIM * OUT_DIM * sizeof(bf16_t));
  float*  feats = (float*)alloc((size_t)G * (OUT_DIM + IN_DIM) * sizeof(float));

  // weight conversion to bf16 (deterministic each call)
  {
    int n1 = NT * OUT_DIM * OUT_DIM;
    cvt_bf16_kernel<<<(n1 + 255) / 256, 256, 0, stream>>>(Ws, Wsb, n1);
    int n2 = 3 * OUT_DIM * OUT_DIM;
    cvt_bf16_kernel<<<(n2 + 255) / 256, 256, 0, stream>>>(Wih, Wihb, n2);
    cvt_bf16_kernel<<<(n2 + 255) / 256, 256, 0, stream>>>(Whh, Whhb, n2);
  }
  {
    size_t n = (size_t)N * OUT_DIM;
    init_h_kernel<<<(unsigned)((n + 255) / 256), 256, 0, stream>>>(nf, hA, hAb, N);
  }

  float*  hc = hA;  bf16_t* hcb = hAb;
  float*  hn = hB;  bf16_t* hnb = hBb;

  for (int step = 0; step < NSTEPS; ++step) {
    size_t nh = (size_t)N * OUT_DIM;
    zero_kernel<<<(unsigned)((nh + 255) / 256), 256, 0, stream>>>(aBuf, nh);

    long tthreads = (long)nRowTiles * NT * 32;               // wave per (rt, t)
    transform_kernel<<<(unsigned)((tthreads + 255) / 256), 256, 0, stream>>>(
        hcb, Wsb, bs, WhBuf, N, nRowTiles);

    long sthreads = (long)E * 32;                            // wave per edge
    scatter_kernel<<<(unsigned)((sthreads + 255) / 256), 256, 0, stream>>>(
        WhBuf, src, dst, etype, aBuf, E, N);

    long gthreads = (long)nRowTiles * 2 * 32;                // wave per (rt, half)
    gru_kernel<<<(unsigned)((gthreads + 255) / 256), 256, 0, stream>>>(
        aBuf, hc, hcb, Wihb, Whhb, bih, bhh, hn, hnb, nRowTiles);

    float*  tf = hc;  hc = hn;   hn = tf;
    bf16_t* tb = hcb; hcb = hnb; hnb = tb;
  }

  {
    size_t nfeat = (size_t)G * (OUT_DIM + IN_DIM);
    zero_kernel<<<(unsigned)((nfeat + 255) / 256), 256, 0, stream>>>(feats, nfeat);
    long athreads = (long)N * 32;
    readout_acc_kernel<<<(unsigned)((athreads + 255) / 256), 256, 0, stream>>>(
        hc, nf, gids, feats, N);
    long fthreads = (long)G * 32;
    readout_final_kernel<<<(unsigned)((fthreads + 255) / 256), 256, 0, stream>>>(
        feats, Wcls, bcls, out, G);
  }
}
